// MPNN3D_5214090297737
// MI455X (gfx1250) — compile-verified, matrix operand order checked by reading, not run on hardware
//
#include <hip/hip_runtime.h>
#include <hip/hip_bf16.h>

#define N_NODES   50000
#define N_EDGES   800000
#define N_GRAPHS  128
#define NODE_DIM  64
#define EDGE_DIM  16
#define HIDDEN    128
#define DEPTH     5
#define RH        512
#define K1        288    // padded concat K: 128 + 128 + 16 + 1(sq) + 1(deg->bias) + 14 pad
#define K2        256
#define MTILES    (N_NODES / 16)   // 3125, exact
#define NTILES    (HIDDEN / 16)    // 8

typedef __bf16 bf16;
typedef __attribute__((ext_vector_type(16))) __bf16 v16bf;
typedef __attribute__((ext_vector_type(8)))  __bf16 v8bf;
typedef __attribute__((ext_vector_type(8)))  float  v8f;

static __device__ inline bf16 tobf(float f) {
    union { float f; unsigned u; } c; c.f = f;
    c.u += 0x7FFFu + ((c.u >> 16) & 1u);           // round-to-nearest-even
    union { unsigned short s; bf16 b; } o; o.s = (unsigned short)(c.u >> 16);
    return o.b;
}

// monotonic float<->uint encoding for atomicMax on floats
static __device__ inline unsigned encf(float f) {
    union { float f; unsigned u; } c; c.f = f;
    return (c.u >> 31) ? ~c.u : (c.u | 0x80000000u);
}
static __device__ inline float decf(unsigned e) {
    union { unsigned u; float f; } c;
    c.u = (e >> 31) ? (e & 0x7FFFFFFFu) : ~e;
    return c.f;
}

// ---------------- conversions (fp32 -> bf16) ----------------
__global__ void k_f32_to_bf16(const float* __restrict__ in, bf16* __restrict__ out, long long n) {
    long long i = (long long)blockIdx.x * blockDim.x + threadIdx.x;
    if (i < n) out[i] = tobf(in[i]);
}

__global__ void k_conv_win(const float* __restrict__ W_in, bf16* __restrict__ WinT) {
    int i = blockIdx.x * blockDim.x + threadIdx.x;
    if (i >= HIDDEN * NODE_DIM) return;
    int n = i / NODE_DIM, k = i % NODE_DIM;
    WinT[n * NODE_DIM + k] = tobf(W_in[k * HIDDEN + n]);
}

__global__ void k_conv_w1(const float* __restrict__ W_pre, const float* __restrict__ b_pre,
                          bf16* __restrict__ W1T) {
    int i = blockIdx.x * blockDim.x + threadIdx.x;
    if (i >= DEPTH * HIDDEN * K1) return;
    int l = i / (HIDDEN * K1);
    int r = i % (HIDDEN * K1);
    int n = r / K1, k = r % K1;
    float v;
    if (k < 273)       v = W_pre[((size_t)l * 273 + k) * HIDDEN + n];
    else if (k == 273) v = b_pre[l * HIDDEN + n];   // multiplied by deg column of X1
    else               v = 0.0f;
    W1T[((size_t)l * HIDDEN + n) * K1 + k] = tobf(v);
}

__global__ void k_conv_w2(const float* __restrict__ W_post, bf16* __restrict__ W2T) {
    int i = blockIdx.x * blockDim.x + threadIdx.x;
    if (i >= DEPTH * HIDDEN * K2) return;
    int l = i / (HIDDEN * K2);
    int r = i % (HIDDEN * K2);
    int n = r / K2, k = r % K2;
    W2T[((size_t)l * HIDDEN + n) * K2 + k] = tobf(W_post[((size_t)l * K2 + k) * HIDDEN + n]);
}

// ---------------- layer-invariant edge scatter: deg, sum(edge_feat), sum(sq) ----------------
__global__ void k_edge_static(const float* __restrict__ x, const float* __restrict__ edge_feat,
                              const int* __restrict__ src, const int* __restrict__ dst,
                              float* __restrict__ deg, float* __restrict__ EF,
                              float* __restrict__ SQ) {
    int e = blockIdx.x * blockDim.x + threadIdx.x;
    if (e >= N_EDGES) return;
    int s = src[e], d = dst[e];
    float dx = x[s * 3 + 0] - x[d * 3 + 0];
    float dy = x[s * 3 + 1] - x[d * 3 + 1];
    float dz = x[s * 3 + 2] - x[d * 3 + 2];
    float sq = dx * dx + dy * dy + dz * dz;
    atomicAdd(&deg[d], 1.0f);
    atomicAdd(&SQ[d], sq);
    const float* ef = edge_feat + (size_t)e * EDGE_DIM;
    float* efd = EF + (size_t)d * EDGE_DIM;
    #pragma unroll
    for (int c = 0; c < EDGE_DIM; ++c) atomicAdd(&efd[c], ef[c]);
}

// ---------------- per-layer scatter: Aag[dst] += h[src] ----------------
__global__ void k_scatter_h(const float* __restrict__ h, const int* __restrict__ src,
                            const int* __restrict__ dst, float* __restrict__ Aag) {
    long long t = (long long)blockIdx.x * blockDim.x + threadIdx.x;
    if (t >= (long long)N_EDGES * 32) return;
    int e  = (int)(t >> 5);
    int ch = ((int)t & 31) * 4;
    int s = src[e], d = dst[e];
    const float4 v = *(const float4*)(h + (size_t)s * HIDDEN + ch);
    float* a = Aag + (size_t)d * HIDDEN + ch;
    atomicAdd(&a[0], v.x);
    atomicAdd(&a[1], v.y);
    atomicAdd(&a[2], v.z);
    atomicAdd(&a[3], v.w);
}

// ---------------- build X1 = [Aag | deg*h | EF | SQ | deg | 0pad] (bf16, N x 288) ----------------
__global__ void k_build_x1(const float* __restrict__ Aag, const float* __restrict__ h,
                           const float* __restrict__ deg, const float* __restrict__ EF,
                           const float* __restrict__ SQ, bf16* __restrict__ X1) {
    long long t = (long long)blockIdx.x * blockDim.x + threadIdx.x;
    if (t >= (long long)N_NODES * K1) return;
    int n = (int)(t / K1), k = (int)(t % K1);
    float v;
    if (k < 128)       v = Aag[(size_t)n * HIDDEN + k];
    else if (k < 256)  v = deg[n] * h[(size_t)n * HIDDEN + (k - 128)];
    else if (k < 272)  v = EF[(size_t)n * EDGE_DIM + (k - 256)];
    else if (k == 272) v = SQ[n];
    else if (k == 273) v = deg[n];
    else               v = 0.0f;
    X1[(size_t)n * K1 + k] = tobf(v);
}

// ---------------- build X2 = [h | m_sum] (bf16, N x 256), written into X1 buffer ----------------
__global__ void k_build_x2(const float* __restrict__ h, const float* __restrict__ msum,
                           bf16* __restrict__ X2) {
    long long t = (long long)blockIdx.x * blockDim.x + threadIdx.x;
    if (t >= (long long)N_NODES * K2) return;
    int n = (int)(t / K2), k = (int)(t % K2);
    float v = (k < 128) ? h[(size_t)n * HIDDEN + k] : msum[(size_t)n * HIDDEN + (k - 128)];
    X2[(size_t)n * K2 + k] = tobf(v);
}

// ---------------- WMMA bf16 GEMM ----------------
// One wave computes one 16-row strip across the full N=128 (8 accumulators).
// A: bf16 [M, K] row-major. WT: bf16 [128, K] (output-column-major weights).
// C[M,128] = act(A @ WT^T + bias + resid)
template <int K, int BIAS, int RESID, int RELU>
__global__ void k_gemm(const bf16* __restrict__ A, const bf16* __restrict__ WT,
                       const float* __restrict__ bias, const float* __restrict__ resid,
                       float* __restrict__ C) {
    int wid = blockIdx.x * (blockDim.x >> 5) + (threadIdx.x >> 5);
    if (wid >= MTILES) return;                 // wave-uniform
    int lane = threadIdx.x & 31;
    int hf   = lane >> 4;                      // half-wave select
    int r    = lane & 15;

    const bf16* arow = A + (size_t)(wid * 16 + r) * K;

    v8f acc[NTILES];
    #pragma unroll
    for (int i = 0; i < NTILES; ++i) acc[i] = (v8f){};

    for (int kk = 0; kk < K; kk += 32) {
        // A fragment: lane<16 -> K kk+0..7 & kk+16..23 ; lane>=16 -> +8 each
        v8bf a0 = *(const v8bf*)(arow + kk + 8 * hf);
        v8bf a1 = *(const v8bf*)(arow + kk + 16 + 8 * hf);
        v16bf a;
        #pragma unroll
        for (int i = 0; i < 8; ++i) { a[i] = a0[i]; a[8 + i] = a1[i]; }
        #pragma unroll
        for (int nt = 0; nt < NTILES; ++nt) {
            const bf16* brow = WT + (size_t)(nt * 16 + r) * K + kk + 8 * hf;
            v8bf b0 = *(const v8bf*)(brow);
            v8bf b1 = *(const v8bf*)(brow + 16);
            v16bf b;
            #pragma unroll
            for (int i = 0; i < 8; ++i) { b[i] = b0[i]; b[8 + i] = b1[i]; }
            acc[nt] = __builtin_amdgcn_wmma_f32_16x16x32_bf16(false, a, false, b,
                                                              (short)0, acc[nt],
                                                              false, false);
        }
    }

    #pragma unroll
    for (int nt = 0; nt < NTILES; ++nt) {
        int col = nt * 16 + r;
        float bv = BIAS ? bias[col] : 0.0f;
        #pragma unroll
        for (int v = 0; v < 8; ++v) {
            int row = wid * 16 + v + 8 * hf;   // C/D layout: VGPR v -> M=v (+8 upper half)
            float val = acc[nt][v] + bv;
            if (RESID) val += resid[(size_t)row * HIDDEN + col];
            if (RELU)  val = fmaxf(val, 0.0f);
            C[(size_t)row * HIDDEN + col] = val;
        }
    }
}

// ---------------- readout ----------------
__global__ void k_init_gmax(unsigned* __restrict__ gmax) {
    int i = blockIdx.x * blockDim.x + threadIdx.x;
    if (i < N_GRAPHS * HIDDEN) gmax[i] = encf(-__builtin_huge_valf());
}

__global__ void k_graph_scatter(const float* __restrict__ h, const int* __restrict__ n2g,
                                float* __restrict__ gsum, unsigned* __restrict__ gmax,
                                float* __restrict__ gcnt) {
    long long t = (long long)blockIdx.x * blockDim.x + threadIdx.x;
    if (t >= (long long)N_NODES * HIDDEN) return;
    int n = (int)(t >> 7), ch = (int)t & 127;
    int g = n2g[n];
    float v = h[(size_t)n * HIDDEN + ch];
    atomicAdd(&gsum[g * HIDDEN + ch], v);
    atomicMax(&gmax[g * HIDDEN + ch], encf(v));
    if (ch == 0) atomicAdd(&gcnt[g], 1.0f);
}

__global__ void k_readout1(const float* __restrict__ gsum, const unsigned* __restrict__ gmax,
                           const float* __restrict__ gcnt, const float* __restrict__ W_r1,
                           const float* __restrict__ b_r1, float* __restrict__ r1) {
    int t = blockIdx.x * blockDim.x + threadIdx.x;
    if (t >= N_GRAPHS * RH) return;
    int g = t / RH, j = t % RH;
    float inv = 1.0f / gcnt[g];
    float acc = b_r1[j];
    for (int k = 0; k < HIDDEN; ++k)
        acc += (gsum[g * HIDDEN + k] * inv) * W_r1[k * RH + j];
    for (int k = 0; k < HIDDEN; ++k)
        acc += decf(gmax[g * HIDDEN + k]) * W_r1[(HIDDEN + k) * RH + j];
    r1[t] = fmaxf(acc, 0.0f);
}

__global__ void k_readout2(const float* __restrict__ r1, const float* __restrict__ W_r2,
                           const float* __restrict__ b_r2, float* __restrict__ out) {
    int g = blockIdx.x * blockDim.x + threadIdx.x;
    if (g >= N_GRAPHS) return;
    float acc = b_r2[0];
    for (int j = 0; j < RH; ++j) acc += r1[g * RH + j] * W_r2[j];
    out[g] = acc;
}

// ---------------- host ----------------
static inline int cdiv(long long a, int b) { return (int)((a + b - 1) / b); }

extern "C" void kernel_launch(void* const* d_in, const int* in_sizes, int n_in,
                              void* d_out, int out_size, void* d_ws, size_t ws_size,
                              hipStream_t stream) {
    (void)in_sizes; (void)n_in; (void)out_size; (void)ws_size;
    const float* node_feat = (const float*)d_in[0];
    const float* x         = (const float*)d_in[1];
    const float* edge_feat = (const float*)d_in[2];
    const int*   src       = (const int*)d_in[3];
    const int*   dst       = (const int*)d_in[4];
    const int*   n2g       = (const int*)d_in[5];
    const float* W_in      = (const float*)d_in[6];
    const float* b_in      = (const float*)d_in[7];
    const float* W_pre     = (const float*)d_in[8];
    const float* b_pre     = (const float*)d_in[9];
    const float* W_post    = (const float*)d_in[10];
    const float* b_post    = (const float*)d_in[11];
    const float* W_r1      = (const float*)d_in[12];
    const float* b_r1      = (const float*)d_in[13];
    const float* W_r2      = (const float*)d_in[14];
    const float* b_r2      = (const float*)d_in[15];
    float* out = (float*)d_out;

    char* base = (char*)d_ws;
    size_t off = 0;
    auto carve = [&](size_t bytes) -> char* {
        char* p = base + off;
        off = (off + bytes + 255) & ~(size_t)255;
        return p;
    };
    float*    h    = (float*)carve((size_t)N_NODES * HIDDEN * 4);
    float*    Aag  = (float*)carve((size_t)N_NODES * HIDDEN * 4);
    float*    msum = (float*)carve((size_t)N_NODES * HIDDEN * 4);
    bf16*     X1   = (bf16*)carve((size_t)N_NODES * K1 * 2);      // reused as X2 (stride 256)
    bf16*     nfb  = (bf16*)carve((size_t)N_NODES * NODE_DIM * 2);
    float*    deg  = (float*)carve((size_t)N_NODES * 4);
    float*    EF   = (float*)carve((size_t)N_NODES * EDGE_DIM * 4);
    float*    SQ   = (float*)carve((size_t)N_NODES * 4);
    bf16*     WinT = (bf16*)carve((size_t)HIDDEN * NODE_DIM * 2);
    bf16*     W1T  = (bf16*)carve((size_t)DEPTH * HIDDEN * K1 * 2);
    bf16*     W2T  = (bf16*)carve((size_t)DEPTH * HIDDEN * K2 * 2);
    float*    gsum = (float*)carve((size_t)N_GRAPHS * HIDDEN * 4);
    unsigned* gmax = (unsigned*)carve((size_t)N_GRAPHS * HIDDEN * 4);
    float*    gcnt = (float*)carve((size_t)N_GRAPHS * 4);
    float*    r1   = (float*)carve((size_t)N_GRAPHS * RH * 4);

    const int B = 256;
    const int gemmGrid = cdiv(MTILES, B / 32);   // 1 wave per 16-row strip

    // static per-node edge aggregates + weight conversion
    hipMemsetAsync(deg, 0, (size_t)N_NODES * 4, stream);
    hipMemsetAsync(EF,  0, (size_t)N_NODES * EDGE_DIM * 4, stream);
    hipMemsetAsync(SQ,  0, (size_t)N_NODES * 4, stream);
    k_conv_win<<<cdiv(HIDDEN * NODE_DIM, B), B, 0, stream>>>(W_in, WinT);
    k_conv_w1<<<cdiv((long long)DEPTH * HIDDEN * K1, B), B, 0, stream>>>(W_pre, b_pre, W1T);
    k_conv_w2<<<cdiv((long long)DEPTH * HIDDEN * K2, B), B, 0, stream>>>(W_post, W2T);
    k_f32_to_bf16<<<cdiv((long long)N_NODES * NODE_DIM, B), B, 0, stream>>>(
        node_feat, nfb, (long long)N_NODES * NODE_DIM);
    k_edge_static<<<cdiv(N_EDGES, B), B, 0, stream>>>(x, edge_feat, src, dst, deg, EF, SQ);

    // input MLP: h = relu(node_feat @ W_in + b_in)
    k_gemm<NODE_DIM, 1, 0, 1><<<gemmGrid, B, 0, stream>>>(nfb, WinT, b_in, nullptr, h);

    for (int l = 0; l < DEPTH; ++l) {
        hipMemsetAsync(Aag, 0, (size_t)N_NODES * HIDDEN * 4, stream);
        k_scatter_h<<<cdiv((long long)N_EDGES * 32, B), B, 0, stream>>>(h, src, dst, Aag);
        k_build_x1<<<cdiv((long long)N_NODES * K1, B), B, 0, stream>>>(Aag, h, deg, EF, SQ, X1);
        // m_sum = X1 @ W1cat[l]   (bias/sq/edge/deg terms folded into X1 columns)
        k_gemm<K1, 0, 0, 0><<<gemmGrid, B, 0, stream>>>(
            X1, W1T + (size_t)l * HIDDEN * K1, nullptr, nullptr, msum);
        k_build_x2<<<cdiv((long long)N_NODES * K2, B), B, 0, stream>>>(h, msum, X1);
        // h = [h | m_sum] @ W_post[l] + b_post[l] + h   (fp32 residual read from h, in place)
        k_gemm<K2, 1, 1, 0><<<gemmGrid, B, 0, stream>>>(
            X1, W2T + (size_t)l * HIDDEN * K2, b_post + l * HIDDEN, h, h);
    }

    // graph readout: mean + max per graph, then 2-layer MLP
    hipMemsetAsync(gsum, 0, (size_t)N_GRAPHS * HIDDEN * 4, stream);
    hipMemsetAsync(gcnt, 0, (size_t)N_GRAPHS * 4, stream);
    k_init_gmax<<<cdiv(N_GRAPHS * HIDDEN, B), B, 0, stream>>>(gmax);
    k_graph_scatter<<<cdiv((long long)N_NODES * HIDDEN, B), B, 0, stream>>>(h, n2g, gsum, gmax, gcnt);
    k_readout1<<<cdiv(N_GRAPHS * RH, B), B, 0, stream>>>(gsum, gmax, gcnt, W_r1, b_r1, r1);
    k_readout2<<<cdiv(N_GRAPHS, B), B, 0, stream>>>(r1, W_r2, b_r2, out);
}
